// EncoderLayer_38165079392901
// MI455X (gfx1250) — compile-verified, matrix operand order checked by reading, not run on hardware
//
#include <hip/hip_runtime.h>
#include <hip/hip_bf16.h>

// ---------------------------------------------------------------------------
// Transformer encoder layer for MI455X (gfx1250, wave32, WMMA).
// B=4, S=4096, D=256, H=4, DEPTH=64, DFF=1024.
// All GEMMs + attention on v_wmma_f32_16x16x32_f16 (f16 in, f32 acc).
// All global->LDS tile staging uses global_load_async_to_lds_b128 with
// double-buffered software pipelining (ASYNCcnt); fragments load as 2x b128.
// ---------------------------------------------------------------------------

#define B_ 4
#define S_ 4096
#define D_ 256
#define H_ 4
#define DEPTH_ 64
#define DFF_ 1024
#define M_ (B_ * S_)

typedef __attribute__((ext_vector_type(16))) _Float16 v16h;
typedef __attribute__((ext_vector_type(8)))  _Float16 v8h;
typedef __attribute__((ext_vector_type(4)))  _Float16 v4h;
typedef __attribute__((ext_vector_type(8)))  float    v8f;
typedef __attribute__((ext_vector_type(4)))  float    v4f;

// Load one 16-half fragment for lane `lane` from a row whose K-run starts at
// `p`.  Lane's data: K offsets [b, b+8) and [b+16, b+24), b = 8*(lane>=16).
__device__ __forceinline__ v16h load_frag(const _Float16* p, int lane) {
    const v8h* q = (const v8h*)(p + ((lane >> 4) << 3));
    v8h lo = q[0];
    v8h hi = q[2];
    return __builtin_shufflevector(lo, hi, 0, 1, 2, 3, 4, 5, 6, 7, 8, 9, 10,
                                   11, 12, 13, 14, 15);
}

// Async 16-byte global -> LDS copy (CDNA5, ASYNCcnt-tracked).
__device__ __forceinline__ void async_copy_b128(_Float16* lds_dst,
                                                const _Float16* gsrc) {
    uint32_t loff = (uint32_t)(uintptr_t)lds_dst;
    asm volatile("global_load_async_to_lds_b128 %0, %1, off"
                 :
                 : "v"(loff), "v"(gsrc)
                 : "memory");
}
__device__ __forceinline__ void wait_async0() {
    asm volatile("s_wait_asynccnt 0" ::: "memory");
}

// ---------------------------------------------------------------------------
// float -> f16 conversion (vectorized 4-wide)
// ---------------------------------------------------------------------------
__global__ void cvt_f16_kernel(const float* __restrict__ src,
                               _Float16* __restrict__ dst, int n4) {
    int i = blockIdx.x * 256 + threadIdx.x;
    if (i < n4) {
        v4f v = ((const v4f*)src)[i];
        ((v4h*)dst)[i] = __builtin_convertvector(v, v4h);
    }
}

// ---------------------------------------------------------------------------
// One-time weight transpose-convert: WT[n][k] (f16) = W[k][n] (f32)
// ---------------------------------------------------------------------------
__global__ void transpose_f16_kernel(const float* __restrict__ W,
                                     _Float16* __restrict__ WT, int K, int N) {
    int n = blockIdx.x * 32 + threadIdx.x;
    int k = blockIdx.y * 8 + threadIdx.y;
    if (n < N && k < K) WT[(size_t)n * K + k] = (_Float16)W[(size_t)k * N + n];
}

// ---------------------------------------------------------------------------
// GEMM: C[M x N] = A[M x K] (f16, row-major) * WT[N x K] (f16, N-major) + bias
// Block: 128 threads (4 waves), 128x64 tile; wave w owns rows w*32..w*32+31.
// Double-buffered async staging.  mode 0: f32 row-major.
// mode 1: f16 split-head [B,H,S,64].  mode 2: f16 split-head T [B,H,64,S].
// ---------------------------------------------------------------------------
__global__ __launch_bounds__(128) void gemm_kernel(
    const _Float16* __restrict__ A, const _Float16* __restrict__ WT,
    const float* __restrict__ bias, float* __restrict__ outF,
    _Float16* __restrict__ outH, int M, int N, int K, int mode, int relu,
    float scale) {
    __shared__ __align__(16) _Float16 As[2][128][32];   // row-major (K-runs)
    __shared__ __align__(16) _Float16 BsT[2][64][32];   // N-major: BsT[n][k]

    const int m0 = blockIdx.x * 128;
    const int n0 = blockIdx.y * 64;
    const int t = threadIdx.x;
    const int wave = t >> 5;
    const int lane = t & 31;

    v8f acc[2][4] = {};

    // stage first k-tile into buffer 0
#pragma unroll
    for (int i = t; i < 512; i += 128) {
        int r = i >> 2, cc = i & 3;
        async_copy_b128(&As[0][r][cc * 8],
                        A + (size_t)(m0 + r) * K + cc * 8);
    }
#pragma unroll
    for (int i = t; i < 256; i += 128) {
        int c = i >> 2, cc = i & 3;
        async_copy_b128(&BsT[0][c][cc * 8],
                        WT + (size_t)(n0 + c) * K + cc * 8);
    }

    for (int k0 = 0; k0 < K; k0 += 32) {
        const int buf = (k0 >> 5) & 1;
        wait_async0();
        __syncthreads();  // tile `buf` ready for all waves

        if (k0 + 32 < K) {  // issue DMA for next tile into other buffer
            const int k1 = k0 + 32;
#pragma unroll
            for (int i = t; i < 512; i += 128) {
                int r = i >> 2, cc = i & 3;
                async_copy_b128(&As[buf ^ 1][r][cc * 8],
                                A + (size_t)(m0 + r) * K + k1 + cc * 8);
            }
#pragma unroll
            for (int i = t; i < 256; i += 128) {
                int c = i >> 2, cc = i & 3;
                async_copy_b128(&BsT[buf ^ 1][c][cc * 8],
                                WT + (size_t)(n0 + c) * K + k1 + cc * 8);
            }
        }

        v16h a0 = load_frag(&As[buf][wave * 32 + (lane & 15)][0], lane);
        v16h a1 = load_frag(&As[buf][wave * 32 + 16 + (lane & 15)][0], lane);
#pragma unroll
        for (int nt = 0; nt < 4; ++nt) {
            v16h bf = load_frag(&BsT[buf][nt * 16 + (lane & 15)][0], lane);
            acc[0][nt] = __builtin_amdgcn_wmma_f32_16x16x32_f16(
                false, a0, false, bf, (short)0, acc[0][nt], false, false);
            acc[1][nt] = __builtin_amdgcn_wmma_f32_16x16x32_f16(
                false, a1, false, bf, (short)0, acc[1][nt], false, false);
        }
        // no trailing barrier: next iteration's wait+barrier protects reuse
    }

    // epilogue
#pragma unroll
    for (int hh = 0; hh < 2; ++hh)
#pragma unroll
        for (int nt = 0; nt < 4; ++nt) {
            int col = n0 + nt * 16 + (lane & 15);
            float bvv = bias[col];
#pragma unroll
            for (int r = 0; r < 8; ++r) {
                int row = m0 + wave * 32 + hh * 16 + r + ((lane >> 4) << 3);
                float v = acc[hh][nt][r] + bvv;
                if (relu) v = fmaxf(v, 0.0f);
                v *= scale;
                int b = row / S_, s = row % S_;
                int h = col >> 6, d = col & 63;
                if (mode == 0) {
                    outF[(size_t)row * N + col] = v;
                } else if (mode == 1) {
                    outH[(((size_t)(b * H_ + h)) * S_ + s) * DEPTH_ + d] =
                        (_Float16)v;
                } else {
                    outH[(((size_t)(b * H_ + h)) * DEPTH_ + d) * S_ + s] =
                        (_Float16)v;
                }
            }
        }
}

// ---------------------------------------------------------------------------
// Flash attention.  q already scaled by 1/sqrt(DEPTH).
// Grid: (S/64, B*H).  Block: 128 threads (4 waves); wave w owns 16 q-rows.
// q/k: [B*H, S, 64] f16.  v: transposed [B*H, 64, S] f16.
// Double-buffered async K/V staging.  ctx out: [B*S, D] f16 (heads merged).
// ---------------------------------------------------------------------------
__global__ __launch_bounds__(128) void attn_kernel(
    const _Float16* __restrict__ qh, const _Float16* __restrict__ kh,
    const _Float16* __restrict__ vT, const float* __restrict__ mask,
    _Float16* __restrict__ ctxh) {
    __shared__ __align__(16) _Float16 Ks[2][64][64];    // [kpos][d]
    __shared__ __align__(16) _Float16 VsT[2][64][64];   // [d][kpos]
    __shared__ __align__(16) _Float16 Ps[4][16][64];    // per-wave P

    const int qt = blockIdx.x;
    const int bh = blockIdx.y;
    const int b = bh / H_;
    const int h = bh % H_;
    const int t = threadIdx.x;
    const int wave = t >> 5;
    const int lane = t & 31;
    const size_t base = (size_t)bh * S_ * DEPTH_;  // q/k and vT share extent
    const int qr0 = qt * 64 + wave * 16;
    const int NT = S_ / 64;

    // Q fragments straight from global (two 16B loads each)
    v16h qA[2];
#pragma unroll
    for (int c = 0; c < 2; ++c)
        qA[c] = load_frag(
            qh + base + (size_t)(qr0 + (lane & 15)) * DEPTH_ + c * 32, lane);

    v8f O[4] = {};
    float mrow[8], lrow[8];
#pragma unroll
    for (int r = 0; r < 8; ++r) { mrow[r] = -1e30f; lrow[r] = 0.0f; }

    // stage first K/V tile into buffer 0
#pragma unroll
    for (int i = t; i < 512; i += 128) {
        int r = i >> 3, cc = i & 7;
        async_copy_b128(&Ks[0][r][cc * 8],
                        kh + base + (size_t)r * DEPTH_ + cc * 8);
        async_copy_b128(&VsT[0][r][cc * 8],
                        vT + base + (size_t)r * S_ + cc * 8);
    }

    for (int kt = 0; kt < NT; ++kt) {
        const int buf = kt & 1;
        wait_async0();
        __syncthreads();  // tile `buf` ready

        if (kt + 1 < NT) {  // DMA next tile into other buffer
            if (t == 0)
                __builtin_prefetch(mask + (size_t)b * S_ + (kt + 1) * 64, 0, 1);
            const int kr = (kt + 1) * 64;
#pragma unroll
            for (int i = t; i < 512; i += 128) {
                int r = i >> 3, cc = i & 7;
                async_copy_b128(
                    &Ks[buf ^ 1][r][cc * 8],
                    kh + base + (size_t)(kr + r) * DEPTH_ + cc * 8);
                async_copy_b128(&VsT[buf ^ 1][r][cc * 8],
                                vT + base + (size_t)r * S_ + kr + cc * 8);
            }
        }

        // logits tile: S = Q @ K^T   (16 x 64)
        v8f sc[4] = {};
#pragma unroll
        for (int nt = 0; nt < 4; ++nt) {
#pragma unroll
            for (int c = 0; c < 2; ++c) {
                v16h bf =
                    load_frag(&Ks[buf][nt * 16 + (lane & 15)][c * 32], lane);
                sc[nt] = __builtin_amdgcn_wmma_f32_16x16x32_f16(
                    false, qA[c], false, bf, (short)0, sc[nt], false, false);
            }
            float mv =
                mask[(size_t)b * S_ + kt * 64 + nt * 16 + (lane & 15)] * -1e9f;
#pragma unroll
            for (int r = 0; r < 8; ++r) sc[nt][r] += mv;
        }

        // online softmax (row groups: lanes 0-15 and 16-31)
        float newm[8];
#pragma unroll
        for (int r = 0; r < 8; ++r) {
            float mx = fmaxf(fmaxf(sc[0][r], sc[1][r]),
                             fmaxf(sc[2][r], sc[3][r]));
#pragma unroll
            for (int off = 1; off < 16; off <<= 1)
                mx = fmaxf(mx, __shfl_xor(mx, off, 32));
            newm[r] = fmaxf(mrow[r], mx);
        }
#pragma unroll
        for (int r = 0; r < 8; ++r) {
            float corr = __expf(mrow[r] - newm[r]);
            float rs = 0.0f;
#pragma unroll
            for (int nt = 0; nt < 4; ++nt) {
                float p = __expf(sc[nt][r] - newm[r]);
                sc[nt][r] = p;
                rs += p;
            }
#pragma unroll
            for (int off = 1; off < 16; off <<= 1)
                rs += __shfl_xor(rs, off, 32);
            lrow[r] = lrow[r] * corr + rs;
            mrow[r] = newm[r];
#pragma unroll
            for (int nt = 0; nt < 4; ++nt) O[nt][r] *= corr;
        }

        // P (C-layout) -> LDS row-major
#pragma unroll
        for (int nt = 0; nt < 4; ++nt)
#pragma unroll
            for (int r = 0; r < 8; ++r)
                Ps[wave][r + ((lane >> 4) << 3)][nt * 16 + (lane & 15)] =
                    (_Float16)sc[nt][r];
        __syncthreads();

        // O += P @ V
#pragma unroll
        for (int c = 0; c < 2; ++c) {
            v16h pa = load_frag(&Ps[wave][lane & 15][c * 32], lane);
#pragma unroll
            for (int nt = 0; nt < 4; ++nt) {
                v16h bv =
                    load_frag(&VsT[buf][nt * 16 + (lane & 15)][c * 32], lane);
                O[nt] = __builtin_amdgcn_wmma_f32_16x16x32_f16(
                    false, pa, false, bv, (short)0, O[nt], false, false);
            }
        }
        // no trailing barrier: next iteration's wait+barrier protects reuse
    }

    // write merged-head context, normalized by l
#pragma unroll
    for (int nt = 0; nt < 4; ++nt)
#pragma unroll
        for (int r = 0; r < 8; ++r) {
            int row = qr0 + r + ((lane >> 4) << 3);
            int d = nt * 16 + (lane & 15);
            ctxh[(size_t)(b * S_ + row) * D_ + h * DEPTH_ + d] =
                (_Float16)(O[nt][r] / lrow[r]);
        }
}

// ---------------------------------------------------------------------------
// LayerNorm over rows of length n (256 or 1024), optional residual add.
// ---------------------------------------------------------------------------
__global__ __launch_bounds__(256) void ln_kernel(
    const float* __restrict__ a, const float* __restrict__ res,
    const float* __restrict__ g, const float* __restrict__ beta, float eps,
    float* __restrict__ outF, _Float16* __restrict__ outH, int n) {
    __shared__ float red[8];
    const int row = blockIdx.x;
    const int t = threadIdx.x;
    const int cnt = n / 256;  // 1 (D) or 4 (DFF)
    const size_t off = (size_t)row * n;

    float vals[4];
    float sum = 0.0f;
    for (int i = 0; i < cnt; ++i) {
        float v = a[off + t + i * 256];
        if (res) v += res[off + t + i * 256];
        vals[i] = v;
        sum += v;
    }
#pragma unroll
    for (int o = 16; o >= 1; o >>= 1) sum += __shfl_xor(sum, o, 32);
    if ((t & 31) == 0) red[t >> 5] = sum;
    __syncthreads();
    sum = 0.0f;
#pragma unroll
    for (int w = 0; w < 8; ++w) sum += red[w];
    const float mu = sum / n;
    __syncthreads();

    float s2 = 0.0f;
    for (int i = 0; i < cnt; ++i) {
        float d = vals[i] - mu;
        s2 += d * d;
    }
#pragma unroll
    for (int o = 16; o >= 1; o >>= 1) s2 += __shfl_xor(s2, o, 32);
    if ((t & 31) == 0) red[t >> 5] = s2;
    __syncthreads();
    s2 = 0.0f;
#pragma unroll
    for (int w = 0; w < 8; ++w) s2 += red[w];
    const float rstd = rsqrtf(s2 / n + eps);

    for (int i = 0; i < cnt; ++i) {
        int c = t + i * 256;
        float y = (vals[i] - mu) * rstd * g[c] + beta[c];
        if (outF) outF[off + c] = y;
        if (outH) outH[off + c] = (_Float16)y;
    }
}

// ---------------------------------------------------------------------------
extern "C" void kernel_launch(void* const* d_in, const int* in_sizes, int n_in,
                              void* d_out, int out_size, void* d_ws,
                              size_t ws_size, hipStream_t stream) {
    const float* x      = (const float*)d_in[0];
    const float* mask   = (const float*)d_in[1];
    const float* wq     = (const float*)d_in[2];
    const float* bq     = (const float*)d_in[3];
    const float* wk     = (const float*)d_in[4];
    const float* bk     = (const float*)d_in[5];
    const float* wv     = (const float*)d_in[6];
    const float* bv     = (const float*)d_in[7];
    const float* wo     = (const float*)d_in[8];
    const float* bo     = (const float*)d_in[9];
    const float* w1     = (const float*)d_in[10];
    const float* b1     = (const float*)d_in[11];
    const float* lnffg  = (const float*)d_in[12];
    const float* lnffb  = (const float*)d_in[13];
    const float* w2     = (const float*)d_in[14];
    const float* b2     = (const float*)d_in[15];
    const float* ln1g   = (const float*)d_in[16];
    const float* ln1b   = (const float*)d_in[17];
    const float* ln2g   = (const float*)d_in[18];
    const float* ln2b   = (const float*)d_in[19];
    float* out = (float*)d_out;

    // workspace carve-up
    char* p = (char*)d_ws;
    _Float16* xh   = (_Float16*)p; p += (size_t)M_ * D_ * 2;    // 8 MiB
    _Float16* qh   = (_Float16*)p; p += (size_t)M_ * D_ * 2;    // 8 MiB
    _Float16* kh   = (_Float16*)p; p += (size_t)M_ * D_ * 2;    // 8 MiB
    _Float16* vh   = (_Float16*)p; p += (size_t)M_ * D_ * 2;    // 8 MiB (vT)
    _Float16* ctxh = (_Float16*)p; p += (size_t)M_ * D_ * 2;    // 8 MiB
    float* tmpF    = (float*)p;    p += (size_t)M_ * D_ * 4;    // 16 MiB
    float* out1F   = (float*)p;    p += (size_t)M_ * D_ * 4;    // 16 MiB
    _Float16* o1h  = (_Float16*)p; p += (size_t)M_ * D_ * 2;    // 8 MiB
    float* hF      = (float*)p;    p += (size_t)M_ * DFF_ * 4;  // 64 MiB
    _Float16* hH   = (_Float16*)p; p += (size_t)M_ * DFF_ * 2;  // 32 MiB
    float* t2F     = (float*)p;    p += (size_t)M_ * D_ * 4;    // 16 MiB
    _Float16* wqT  = (_Float16*)p; p += (size_t)D_ * D_ * 2;
    _Float16* wkT  = (_Float16*)p; p += (size_t)D_ * D_ * 2;
    _Float16* wvT  = (_Float16*)p; p += (size_t)D_ * D_ * 2;
    _Float16* woT  = (_Float16*)p; p += (size_t)D_ * D_ * 2;
    _Float16* w1T  = (_Float16*)p; p += (size_t)D_ * DFF_ * 2;
    _Float16* w2T  = (_Float16*)p; p += (size_t)DFF_ * D_ * 2;

    // one-time weight transpose-convert (f32 KxN -> f16 NxK)
    dim3 tb(32, 8);
    transpose_f16_kernel<<<dim3(D_ / 32, D_ / 8), tb, 0, stream>>>(wq, wqT, D_, D_);
    transpose_f16_kernel<<<dim3(D_ / 32, D_ / 8), tb, 0, stream>>>(wk, wkT, D_, D_);
    transpose_f16_kernel<<<dim3(D_ / 32, D_ / 8), tb, 0, stream>>>(wv, wvT, D_, D_);
    transpose_f16_kernel<<<dim3(D_ / 32, D_ / 8), tb, 0, stream>>>(wo, woT, D_, D_);
    transpose_f16_kernel<<<dim3(DFF_ / 32, D_ / 8), tb, 0, stream>>>(w1, w1T, D_, DFF_);
    transpose_f16_kernel<<<dim3(D_ / 32, DFF_ / 8), tb, 0, stream>>>(w2, w2T, DFF_, D_);

    // x -> f16
    cvt_f16_kernel<<<(M_ * D_ / 4 + 255) / 256, 256, 0, stream>>>(
        x, xh, M_ * D_ / 4);

    // q/k/v projections (q pre-scaled by 1/sqrt(DEPTH); v written transposed)
    dim3 gQKV(M_ / 128, D_ / 64);
    gemm_kernel<<<gQKV, 128, 0, stream>>>(xh, wqT, bq, nullptr, qh, M_, D_, D_,
                                          1, 0, 0.125f);
    gemm_kernel<<<gQKV, 128, 0, stream>>>(xh, wkT, bk, nullptr, kh, M_, D_, D_,
                                          1, 0, 1.0f);
    gemm_kernel<<<gQKV, 128, 0, stream>>>(xh, wvT, bv, nullptr, vh, M_, D_, D_,
                                          2, 0, 1.0f);

    // flash attention
    attn_kernel<<<dim3(S_ / 64, B_ * H_), 128, 0, stream>>>(qh, kh, vh, mask,
                                                            ctxh);

    // output projection + residual LN (eps 1e-3)
    gemm_kernel<<<gQKV, 128, 0, stream>>>(ctxh, woT, bo, tmpF, nullptr, M_, D_,
                                          D_, 0, 0, 1.0f);
    ln_kernel<<<M_, 256, 0, stream>>>(tmpF, x, ln1g, ln1b, 1e-3f, out1F, o1h,
                                      D_);

    // FFN: relu(out1@w1+b1) -> LN(eps 1e-6) -> @w2+b2
    gemm_kernel<<<dim3(M_ / 128, DFF_ / 64), 128, 0, stream>>>(
        o1h, w1T, b1, hF, nullptr, M_, DFF_, D_, 0, 1, 1.0f);
    ln_kernel<<<M_, 256, 0, stream>>>(hF, nullptr, lnffg, lnffb, 1e-6f,
                                      nullptr, hH, DFF_);
    gemm_kernel<<<dim3(M_ / 128, D_ / 64), 128, 0, stream>>>(
        hH, w2T, b2, t2F, nullptr, M_, D_, DFF_, 0, 0, 1.0f);

    // final residual LN (eps 1e-3) -> d_out
    ln_kernel<<<M_, 256, 0, stream>>>(t2F, out1F, ln2g, ln2b, 1e-3f, out,
                                      nullptr, D_);
}
